// BahdanauAttention_72018011619862
// MI455X (gfx1250) — matrix operand order repaired and checked by reading, not compiled, not run on hardware
//
#include <hip/hip_runtime.h>
#include <math.h>

// Problem sizes (fixed by the reference)
#define B_ 32
#define S_ 2048
#define H_ 1024

typedef __attribute__((ext_vector_type(16))) __bf16        v16bf;
typedef __attribute__((ext_vector_type(8)))  float         v8f;
typedef __attribute__((ext_vector_type(4)))  unsigned int  v4u;
typedef __attribute__((ext_vector_type(4)))  float         v4f;

union FragU { v4u u[2]; v16bf v; };

__device__ __forceinline__ unsigned short f32_to_bf16(float f) {
  unsigned int u = __builtin_bit_cast(unsigned int, f);
  unsigned int r = u + 0x7FFFu + ((u >> 16) & 1u);   // round-to-nearest-even
  return (unsigned short)(r >> 16);
}

// Branch-free tanh: prefer the CDNA5 hardware transcendental v_tanh_f32,
// else build from v_exp_f32 + v_rcp_f32 (no EXEC divergence either way).
__device__ __forceinline__ float fast_tanh(float x) {
#if __has_builtin(__builtin_amdgcn_tanhf)
  return __builtin_amdgcn_tanhf(x);
#else
  float ax = __builtin_fabsf(x);
  // e^(-2|x|) = 2^(-2|x| * log2(e))
  float t  = __builtin_amdgcn_exp2f(ax * -2.885390081777927f);
  float r  = (1.0f - t) * __builtin_amdgcn_rcpf(1.0f + t);
  return __builtin_copysignf(r, x);
#endif
}

// ---------------------------------------------------------------------------
// Kernel 1: convert Ua_w (H x H fp32, row-major [out][in]) to bf16 workspace.
// ---------------------------------------------------------------------------
__global__ void ua_convert_kernel(const float* __restrict__ Ua,
                                  unsigned short* __restrict__ out) {
  int i = (blockIdx.x * 256 + threadIdx.x) * 4;
  v4f f = *(const v4f*)(Ua + i);
  unsigned int lo = (unsigned int)f32_to_bf16(f.x) | ((unsigned int)f32_to_bf16(f.y) << 16);
  unsigned int hi = (unsigned int)f32_to_bf16(f.z) | ((unsigned int)f32_to_bf16(f.w) << 16);
  *(uint2*)(out + i) = make_uint2(lo, hi);
}

// ---------------------------------------------------------------------------
// Kernel 2: q_proj[b][o] = query[b,:] . Wa_w[o,:] + Wa_b[o]   (tiny GEMM, fp32)
// ---------------------------------------------------------------------------
__global__ void qproj_kernel(const float* __restrict__ query,
                             const float* __restrict__ Wa_w,
                             const float* __restrict__ Wa_b,
                             float* __restrict__ qproj) {
  __shared__ float qrow[H_];
  int b = blockIdx.y;
  int o = blockIdx.x * 256 + threadIdx.x;
  for (int h = threadIdx.x; h < H_; h += 256) qrow[h] = query[b * H_ + h];
  __syncthreads();
  const float* wr = Wa_w + (size_t)o * H_;
  float acc = Wa_b[o];
  for (int h = 0; h < H_; h += 4) {
    v4f w = *(const v4f*)(wr + h);
    acc += qrow[h] * w.x + qrow[h + 1] * w.y + qrow[h + 2] * w.z + qrow[h + 3] * w.w;
  }
  qproj[b * H_ + o] = acc;
}

// ---------------------------------------------------------------------------
// Kernel 3: fused k_proj GEMM (bf16 WMMA) + bias + tanh + Va dot -> scores
// Grid: (S_/128, B_), 256 threads (8 wave32s). Each block:
//   - stages keys[b, s0:s0+128, 0:1024] as bf16 in LDS (258 KB, full K extent;
//     only possible thanks to CDNA5's 320 KB WGP LDS -> no K-split re-reads)
//   - loops N in chunks of 64; for each chunk accumulates K=1024 via
//     v_wmma_f32_16x16x32_bf16, then fused bias+tanh+Va epilogue per tile
// ---------------------------------------------------------------------------
#define LDSP (H_ + 8)   // bf16 row pitch: +16B so row stride = 516 dwords
                        // -> lane l hits banks {4l..4l+3}: conflict-free b128

__global__ void score_kernel(const float* __restrict__ keys,
                             const unsigned short* __restrict__ ua16,
                             const float* __restrict__ qproj,
                             const float* __restrict__ Ua_b,
                             const float* __restrict__ Va_w,
                             const float* __restrict__ Va_b,
                             float* __restrict__ scores) {
  extern __shared__ unsigned short As[];   // [128][LDSP] bf16

  const int b   = blockIdx.y;
  const int s0  = blockIdx.x * 128;
  const int tid = threadIdx.x;

  // ---- stage & convert keys tile: 128 x 1024 fp32 -> bf16 LDS ----
  const float* kbase = keys + ((size_t)b * S_ + s0) * H_;
  for (int it = 0; it < 128; ++it) {
    int idx = it * 1024 + tid * 4;         // 256 thr * 4 = 1024 elems / iter
    v4f f = *(const v4f*)(kbase + idx);
    int r = idx >> 10;
    int h = idx & (H_ - 1);
    unsigned int lo = (unsigned int)f32_to_bf16(f.x) | ((unsigned int)f32_to_bf16(f.y) << 16);
    unsigned int hi = (unsigned int)f32_to_bf16(f.z) | ((unsigned int)f32_to_bf16(f.w) << 16);
    *(uint2*)(As + r * LDSP + h) = make_uint2(lo, hi);
  }
  __syncthreads();

  const int wave  = tid >> 5;
  const int lane  = tid & 31;
  const int lhalf = (lane < 16) ? 0 : 1;   // which 16-lane half
  const int lm    = lane & 15;

  // A-fragment source row in LDS (16-bit A 16x32 layout: lanes 0-15 = rows,
  // lanes 16-31 same rows but K+8 / K+24 halves)
  const unsigned short* Arow = As + (wave * 16 + lm) * LDSP;

  float psum[8];
#pragma unroll
  for (int g = 0; g < 8; ++g) psum[g] = 0.0f;

  for (int nc = 0; nc < H_; nc += 64) {    // 16 N-chunks of 64 columns
    v8f acc[4];
#pragma unroll
    for (int t = 0; t < 4; ++t)
#pragma unroll
      for (int g = 0; g < 8; ++g) acc[t][g] = 0.0f;

    for (int k0 = 0; k0 < H_; k0 += 32) {  // full-K accumulation
      // A fragment: K = k0 + {0..7, 16..23} (+8 for high half lanes)
      int ka = k0 + lhalf * 8;
      FragU a;
      a.u[0] = *(const v4u*)(Arow + ka);
      a.u[1] = *(const v4u*)(Arow + ka + 16);
#pragma unroll
      for (int t = 0; t < 4; ++t) {
        // B fragment: column n = Ua row n; K = k0 + {0..7, 16..23} (+8 for high half lanes)
        int n = nc + t * 16 + lm;
        const unsigned short* brow = ua16 + (size_t)n * H_ + k0 + lhalf * 8;
        FragU bf;
        bf.u[0] = *(const v4u*)(brow);
        bf.u[1] = *(const v4u*)(brow + 16);
        acc[t] = __builtin_amdgcn_wmma_f32_16x16x32_bf16(
            false, a.v, false, bf.v, (short)0, acc[t], false, false);
      }
    }

    // epilogue: e = tanh(qproj + Ua_b + kproj); psum += e * Va  (branch-free)
#pragma unroll
    for (int t = 0; t < 4; ++t) {
      int n = nc + t * 16 + lm;            // C layout: lane half -> same N index
      float qp = qproj[b * H_ + n] + Ua_b[n];
      float va = Va_w[n];
#pragma unroll
      for (int g = 0; g < 8; ++g) {        // C layout: VGPR g -> M = lhalf*8+g
        float e = fast_tanh(qp + acc[t][g]);
        psum[g] += e * va;
      }
    }
  }

  // reduce over the 16 lanes of each half-group (covers all N)
#pragma unroll
  for (int g = 0; g < 8; ++g) {
    float v = psum[g];
    v += __shfl_xor(v, 8, 16);
    v += __shfl_xor(v, 4, 16);
    v += __shfl_xor(v, 2, 16);
    v += __shfl_xor(v, 1, 16);
    psum[g] = v;
  }
  if (lm == 0) {
    float vb = Va_b[0];
#pragma unroll
    for (int g = 0; g < 8; ++g) {
      int m = lhalf * 8 + g;
      scores[b * S_ + s0 + wave * 16 + m] = psum[g] + vb;
    }
  }
}

// ---------------------------------------------------------------------------
// Kernel 4: row softmax over S (one block per batch row)
// ---------------------------------------------------------------------------
__global__ void softmax_kernel(const float* __restrict__ scores,
                               float* __restrict__ weights) {
  __shared__ float red[256];
  int b = blockIdx.x, tid = threadIdx.x;
  float vals[8];
  float m = -3.4e38f;
#pragma unroll
  for (int i = 0; i < 8; ++i) {
    vals[i] = scores[b * S_ + tid + i * 256];
    m = fmaxf(m, vals[i]);
  }
  red[tid] = m;
  __syncthreads();
  for (int off = 128; off > 0; off >>= 1) {
    if (tid < off) red[tid] = fmaxf(red[tid], red[tid + off]);
    __syncthreads();
  }
  float gmax = red[0];
  __syncthreads();
  float s = 0.0f;
#pragma unroll
  for (int i = 0; i < 8; ++i) {
    vals[i] = __expf(vals[i] - gmax);
    s += vals[i];
  }
  red[tid] = s;
  __syncthreads();
  for (int off = 128; off > 0; off >>= 1) {
    if (tid < off) red[tid] += red[tid + off];
    __syncthreads();
  }
  float inv = 1.0f / red[0];
#pragma unroll
  for (int i = 0; i < 8; ++i)
    weights[b * S_ + tid + i * 256] = vals[i] * inv;
}

// ---------------------------------------------------------------------------
// Kernel 5: context[b][h] = sum_s weights[b][s] * keys[b][s][h]
// ---------------------------------------------------------------------------
__global__ void context_kernel(const float* __restrict__ keys,
                               const float* __restrict__ weights,
                               float* __restrict__ ctx) {
  __shared__ float w[S_];                  // 8 KB
  int b = blockIdx.y;
  int h = blockIdx.x * 256 + threadIdx.x;
  for (int s = threadIdx.x; s < S_; s += 256) w[s] = weights[b * S_ + s];
  __syncthreads();
  const float* kb = keys + (size_t)b * S_ * H_ + h;
  float acc = 0.0f;
  for (int s = 0; s < S_; s += 8) {
    __builtin_prefetch(kb + (size_t)(s + 16) * H_, 0, 1);  // global_prefetch_b8
#pragma unroll
    for (int j = 0; j < 8; ++j)
      acc += w[s + j] * kb[(size_t)(s + j) * H_];
  }
  ctx[b * H_ + h] = acc;
}

// ---------------------------------------------------------------------------
// Launch
// ---------------------------------------------------------------------------
extern "C" void kernel_launch(void* const* d_in, const int* in_sizes, int n_in,
                              void* d_out, int out_size, void* d_ws, size_t ws_size,
                              hipStream_t stream) {
  (void)in_sizes; (void)n_in; (void)out_size; (void)ws_size;
  const float* query = (const float*)d_in[0];
  const float* keys  = (const float*)d_in[1];
  const float* Wa_w  = (const float*)d_in[2];
  const float* Wa_b  = (const float*)d_in[3];
  const float* Ua_w  = (const float*)d_in[4];
  const float* Ua_b  = (const float*)d_in[5];
  const float* Va_w  = (const float*)d_in[6];
  const float* Va_b  = (const float*)d_in[7];

  float* out_ctx = (float*)d_out;                 // [B,H]
  float* out_w   = (float*)d_out + B_ * H_;       // [B,S]

  // workspace layout: bf16 Ua (2MB) | q_proj (128KB) | scores (256KB)
  char* ws = (char*)d_ws;
  unsigned short* ua16 = (unsigned short*)ws;
  float* qproj  = (float*)(ws + (size_t)H_ * H_ * 2);
  float* scores = (float*)(ws + (size_t)H_ * H_ * 2 + (size_t)B_ * H_ * 4);

  ua_convert_kernel<<<(H_ * H_) / 1024, 256, 0, stream>>>(Ua_w, ua16);
  qproj_kernel<<<dim3(H_ / 256, B_), 256, 0, stream>>>(query, Wa_w, Wa_b, qproj);

  size_t smem = (size_t)128 * LDSP * sizeof(unsigned short);  // 258 KB < 320 KB WGP LDS
  hipFuncSetAttribute(reinterpret_cast<const void*>(&score_kernel),
                      hipFuncAttributeMaxDynamicSharedMemorySize, (int)smem);
  score_kernel<<<dim3(S_ / 128, B_), 256, smem, stream>>>(
      keys, ua16, qproj, Ua_b, Va_w, Va_b, scores);

  softmax_kernel<<<B_, 256, 0, stream>>>(scores, out_w);
  context_kernel<<<dim3(H_ / 256, B_), 256, 0, stream>>>(keys, out_w, out_ctx);
}
